// BiomechanicGATHead_53352083751433
// MI455X (gfx1250) — matrix operand order repaired and checked by reading, not compiled, not run on hardware
//
#include <hip/hip_runtime.h>
#include <hip/hip_bf16.h>
#include <math.h>

// ---------------------------------------------------------------------------
// BiomechanicGATHead on MI455X (gfx1250), wave32 + WMMA f32_16x16x32_f16.
// Fully fused: one wave owns a 16-token tile; all intermediates in LDS.
// Round 3: GELU uses v_rcp_f32 (__builtin_amdgcn_rcpf) instead of the IEEE
// divide sequence (div_scale/div_fmas/div_fixup) seen in round-2 asm.
// ---------------------------------------------------------------------------

typedef __attribute__((ext_vector_type(16))) _Float16 v16h;
typedef __attribute__((ext_vector_type(8)))  float    v8f;

#define D_MODEL 128
#define HID     256
#define N_NODES 17
#define NODE_DIM 32
#define NODEF   (N_NODES * NODE_DIM)   // 544
#define TOKENS  (16 * 4096)            // 65536

union F16Frag { v16h v; _Float16 h[16]; };

// Branch-free erf-GELU, Abramowitz-Stegun 7.1.26 (|err(erf)| <= 1.5e-7).
// t = 1/(1 + 0.2316419|x|) via single v_rcp_f32 (approx error << approx poly).
// exp(-x^2/2) via native v_exp_f32 (TRANS pipe, co-executes with WMMA).
__device__ __forceinline__ float gelu_erf(float x) {
    float u  = fabsf(x);
    float t  = __builtin_amdgcn_rcpf(fmaf(0.2316419f, u, 1.0f));
    float p  = fmaf(1.061405429f, t, -1.453152027f);
    p = fmaf(p, t, 1.421413741f);
    p = fmaf(p, t, -0.284496736f);
    p = fmaf(p, t, 0.254829592f);
    p = p * t;
    float e    = __expf(-0.5f * u * u);
    float erfv = copysignf(fmaf(-p, e, 1.0f), x);
    float hx   = 0.5f * x;
    return fmaf(hx, erfv, hx);
}

// ---------------------------------------------------------------------------
// Pack a KxN fp32 weight into WMMA B-fragment order, f16.
// B layout (16x16x32, wave32): lane l -> column (l&15); lanes 0-15 hold
// K 0..7,16..23 of the k-step, lanes 16-31 hold K 8..15,24..31. One fragment = 16
// contiguous halves per lane -> a single 32B load in the main kernel.
// dst[((nt*KS + s)*32 + l)*16 + j] = W[s*32 + 8*(l>>4) + (j&7) + 16*(j>>3)][nt*16 + (l&15)]
// ---------------------------------------------------------------------------
__global__ void pack_b_kernel(const float* __restrict__ W, _Float16* __restrict__ dst,
                              int K, int N, int NT, int KS) {
    int total = NT * KS * 32 * 16;
    for (int idx = blockIdx.x * blockDim.x + threadIdx.x; idx < total;
         idx += gridDim.x * blockDim.x) {
        int j    = idx & 15;
        int l    = (idx >> 4) & 31;
        int frag = idx >> 9;
        int s    = frag % KS;
        int nt   = frag / KS;
        int k    = s * 32 + ((l >> 4) << 3) + (j & 7) + ((j >> 3) << 4);
        int col  = nt * 16 + (l & 15);
        float v  = (k < K && col < N) ? W[(size_t)k * N + col] : 0.0f;
        dst[idx] = (_Float16)v;
    }
}

// Row softmax of the two 17x17 adjacency logit matrices.
__global__ void softmax_adj_kernel(const float* __restrict__ a1, const float* __restrict__ a2,
                                   float* __restrict__ o1, float* __restrict__ o2) {
    int r = threadIdx.x;
    if (r >= 2 * N_NODES) return;
    const float* a = (r < N_NODES) ? a1 : a2;
    float*       o = (r < N_NODES) ? o1 : o2;
    int row = (r < N_NODES) ? r : r - N_NODES;
    float m = -1e30f;
    for (int j = 0; j < N_NODES; ++j) m = fmaxf(m, a[row * N_NODES + j]);
    float e[N_NODES]; float s = 0.f;
    for (int j = 0; j < N_NODES; ++j) { e[j] = expf(a[row * N_NODES + j] - m); s += e[j]; }
    float inv = 1.0f / s;
    for (int j = 0; j < N_NODES; ++j) o[row * N_NODES + j] = e[j] * inv;
}

// ---------------------------------------------------------------------------
// GAT layer (per 16-token tile, single wave):
//   bufTmp = softmax(adj) @ bufRes        (VALU, tiny: 17x17 mix per token)
//   bufRes = gelu(bufTmp @ Wg + bg) + bufRes   (WMMA over 272x32x32)
// Buffers are [16 tokens][17 nodes][32 feat] f16, viewed as [272][32].
// ---------------------------------------------------------------------------
__device__ __forceinline__ void gat_layer(_Float16* __restrict__ bufRes,
                                          _Float16* __restrict__ bufTmp,
                                          const float* __restrict__ sAdj,
                                          const _Float16* __restrict__ pWg,
                                          const float* __restrict__ bg,
                                          int lane) {
    int mrow = lane & 15, khalf = lane >> 4;

    // graph mix: lane = feature channel (32 lanes == 32 features)
    #pragma unroll 1
    for (int t = 0; t < 16; ++t) {
        float nj[N_NODES];
        #pragma unroll
        for (int j = 0; j < N_NODES; ++j)
            nj[j] = (float)bufRes[t * NODEF + j * NODE_DIM + lane];
        #pragma unroll 1
        for (int i = 0; i < N_NODES; ++i) {
            float acc = 0.f;
            #pragma unroll
            for (int j = 0; j < N_NODES; ++j)
                acc = fmaf(sAdj[i * N_NODES + j], nj[j], acc);
            bufTmp[t * NODEF + i * NODE_DIM + lane] = (_Float16)acc;
        }
    }

    // per-node linear + GELU + residual: 272 rows, K=32, N=32
    #pragma unroll 1
    for (int mt = 0; mt < 17; ++mt) {
        F16Frag am;
        int r = mt * 16 + mrow;
        #pragma unroll
        for (int j = 0; j < 8; ++j) am.h[j]     = bufTmp[r * NODE_DIM + khalf * 8 + j];
        #pragma unroll
        for (int j = 0; j < 8; ++j) am.h[8 + j] = bufTmp[r * NODE_DIM + 16 + khalf * 8 + j];
        #pragma unroll
        for (int nt = 0; nt < 2; ++nt) {
            v8f acc = {};
            v16h b = *(const v16h*)(pWg + ((size_t)(nt * 32 + lane)) * 16);
            acc = __builtin_amdgcn_wmma_f32_16x16x32_f16(false, am.v, false, b,
                                                         (short)0, acc, false, false);
            int col = nt * 16 + mrow;
            float bias = bg[col];
            #pragma unroll
            for (int v = 0; v < 8; ++v) {
                int r2 = mt * 16 + khalf * 8 + v;
                float res = (float)bufRes[r2 * NODE_DIM + col];
                bufRes[r2 * NODE_DIM + col] = (_Float16)(gelu_erf(acc[v] + bias) + res);
            }
        }
    }
}

// ---------------------------------------------------------------------------
// Main fused kernel: 128 threads = 4 waves, each wave owns 16 tokens.
// LDS per wave: bufA[16][544] f16 + bufB[16][544] f16 (34816 B).
// ---------------------------------------------------------------------------
__global__ __launch_bounds__(128)
void gat_head_kernel(const float* __restrict__ x,
                     const _Float16* __restrict__ pW1, const float* __restrict__ b1,
                     const _Float16* __restrict__ pW2, const float* __restrict__ b2,
                     const float* __restrict__ adj1s, const _Float16* __restrict__ pWg1,
                     const float* __restrict__ bg1,
                     const float* __restrict__ adj2s, const _Float16* __restrict__ pWg2,
                     const float* __restrict__ bg2,
                     const _Float16* __restrict__ pWc, const float* __restrict__ bc,
                     float* __restrict__ out) {
    extern __shared__ char smem[];
    const int WAVE_BYTES = 2 * 16 * NODEF * (int)sizeof(_Float16);  // 34816
    int wid  = threadIdx.x >> 5;
    int lane = threadIdx.x & 31;
    _Float16* bufA = (_Float16*)(smem + (size_t)wid * WAVE_BYTES);
    _Float16* bufB = bufA + 16 * NODEF;
    float* sAdj1 = (float*)(smem + 4 * (size_t)WAVE_BYTES);
    float* sAdj2 = sAdj1 + N_NODES * N_NODES;

    for (int i = threadIdx.x; i < N_NODES * N_NODES; i += blockDim.x) {
        sAdj1[i] = adj1s[i];
        sAdj2[i] = adj2s[i];
    }
    __syncthreads();

    int mrow  = lane & 15;   // A-layout: lane holds matrix row (l&15)
    int khalf = lane >> 4;   // lanes 16-31 hold the +8 / +16 K offsets
    size_t tileTok = ((size_t)blockIdx.x * 4 + wid) * 16;

    // ---- stage 1: h = gelu(x @ W1 + b1) -> bufB[16][256] f16 --------------
    // A fragments straight from HBM (two 8-float contiguous runs per lane).
    F16Frag ax[4];
    {
        const float* xr = x + (tileTok + mrow) * D_MODEL;
        #pragma unroll
        for (int s = 0; s < 4; ++s) {
            int k0 = s * 32 + khalf * 8;
            #pragma unroll
            for (int j = 0; j < 8; ++j) ax[s].h[j]     = (_Float16)xr[k0 + j];
            #pragma unroll
            for (int j = 0; j < 8; ++j) ax[s].h[8 + j] = (_Float16)xr[k0 + 16 + j];
        }
    }
    #pragma unroll 1
    for (int nt = 0; nt < 16; ++nt) {
        v8f acc = {};
        #pragma unroll
        for (int s = 0; s < 4; ++s) {
            v16h b = *(const v16h*)(pW1 + (((size_t)(nt * 4 + s)) * 32 + lane) * 16);
            acc = __builtin_amdgcn_wmma_f32_16x16x32_f16(false, ax[s].v, false, b,
                                                         (short)0, acc, false, false);
        }
        int col = nt * 16 + mrow;        // C layout: lane&15 = N column
        float bias = b1[col];
        #pragma unroll
        for (int v = 0; v < 8; ++v)
            bufB[(khalf * 8 + v) * HID + col] = (_Float16)gelu_erf(acc[v] + bias);
    }

    // ---- stage 2: nodes = h @ W2 + b2 -> bufA[16][544] --------------------
    F16Frag ah[8];
    #pragma unroll
    for (int s = 0; s < 8; ++s) {
        int k0 = s * 32 + khalf * 8;
        #pragma unroll
        for (int j = 0; j < 8; ++j) ah[s].h[j]     = bufB[mrow * HID + k0 + j];
        #pragma unroll
        for (int j = 0; j < 8; ++j) ah[s].h[8 + j] = bufB[mrow * HID + k0 + 16 + j];
    }
    #pragma unroll 1
    for (int nt = 0; nt < 34; ++nt) {
        v8f acc = {};
        #pragma unroll
        for (int s = 0; s < 8; ++s) {
            v16h b = *(const v16h*)(pW2 + (((size_t)(nt * 8 + s)) * 32 + lane) * 16);
            acc = __builtin_amdgcn_wmma_f32_16x16x32_f16(false, ah[s].v, false, b,
                                                         (short)0, acc, false, false);
        }
        int col = nt * 16 + mrow;
        float bias = b2[col];
        #pragma unroll
        for (int v = 0; v < 8; ++v)
            bufA[(khalf * 8 + v) * NODEF + col] = (_Float16)(acc[v] + bias);
    }

    // ---- stages 3/4: two GAT layers --------------------------------------
    gat_layer(bufA, bufB, sAdj1, pWg1, bg1, lane);
    gat_layer(bufA, bufB, sAdj2, pWg2, bg2, lane);

    // ---- stage 5: out = nodes @ Wc + bc  (N padded 2 -> 16) --------------
    float bc0 = bc[0], bc1 = bc[1];
    #pragma unroll 1
    for (int mt = 0; mt < 17; ++mt) {
        F16Frag am;
        int r = mt * 16 + mrow;
        #pragma unroll
        for (int j = 0; j < 8; ++j) am.h[j]     = bufA[r * NODE_DIM + khalf * 8 + j];
        #pragma unroll
        for (int j = 0; j < 8; ++j) am.h[8 + j] = bufA[r * NODE_DIM + 16 + khalf * 8 + j];
        v8f acc = {};
        v16h b = *(const v16h*)(pWc + (size_t)lane * 16);
        acc = __builtin_amdgcn_wmma_f32_16x16x32_f16(false, am.v, false, b,
                                                     (short)0, acc, false, false);
        int col = mrow;                  // only columns 0,1 are real
        if (col < 2) {
            float bias = (col == 0) ? bc0 : bc1;
            #pragma unroll
            for (int v = 0; v < 8; ++v) {
                int r2 = mt * 16 + khalf * 8 + v;   // r2 = t*17 + node
                out[tileTok * 34 + (size_t)r2 * 2 + col] = acc[v] + bias;
            }
        }
    }
}

// ---------------------------------------------------------------------------
extern "C" void kernel_launch(void* const* d_in, const int* in_sizes, int n_in,
                              void* d_out, int out_size, void* d_ws, size_t ws_size,
                              hipStream_t stream) {
    const float* x    = (const float*)d_in[0];
    const float* W1   = (const float*)d_in[1];
    const float* b1   = (const float*)d_in[2];
    const float* W2   = (const float*)d_in[3];
    const float* b2   = (const float*)d_in[4];
    const float* adj1 = (const float*)d_in[5];
    const float* Wg1  = (const float*)d_in[6];
    const float* bg1  = (const float*)d_in[7];
    const float* adj2 = (const float*)d_in[8];
    const float* Wg2  = (const float*)d_in[9];
    const float* bg2  = (const float*)d_in[10];
    const float* Wc   = (const float*)d_in[11];
    const float* bc   = (const float*)d_in[12];
    float* out = (float*)d_out;

    // workspace carve-up (256B aligned)
    char* ws = (char*)d_ws;
    size_t off = 0;
    auto carve = [&](size_t bytes) {
        char* p = ws + off;
        off = (off + bytes + 255) & ~(size_t)255;
        return p;
    };
    _Float16* pW1  = (_Float16*)carve((size_t)16 * 4 * 32 * 16 * sizeof(_Float16)); // 64 KB
    _Float16* pW2  = (_Float16*)carve((size_t)34 * 8 * 32 * 16 * sizeof(_Float16)); // 272 KB
    _Float16* pWg1 = (_Float16*)carve((size_t)2 * 1 * 32 * 16 * sizeof(_Float16));
    _Float16* pWg2 = (_Float16*)carve((size_t)2 * 1 * 32 * 16 * sizeof(_Float16));
    _Float16* pWc  = (_Float16*)carve((size_t)1 * 1 * 32 * 16 * sizeof(_Float16));
    float*    sA1  = (float*)carve((size_t)N_NODES * N_NODES * sizeof(float));
    float*    sA2  = (float*)carve((size_t)N_NODES * N_NODES * sizeof(float));

    // weight repack (fragment order) + adjacency softmax
    pack_b_kernel<<<64, 256, 0, stream>>>(W1, pW1, 128, 256, 16, 4);
    pack_b_kernel<<<272, 256, 0, stream>>>(W2, pW2, 256, 544, 34, 8);
    pack_b_kernel<<<4, 256, 0, stream>>>(Wg1, pWg1, 32, 32, 2, 1);
    pack_b_kernel<<<4, 256, 0, stream>>>(Wg2, pWg2, 32, 32, 2, 1);
    pack_b_kernel<<<2, 256, 0, stream>>>(Wc, pWc, 32, 2, 1, 1);
    softmax_adj_kernel<<<1, 64, 0, stream>>>(adj1, adj2, sA1, sA2);

    // main fused kernel: 65536 tokens / (16 tok/wave * 4 waves) = 1024 blocks
    size_t shmem = 4 * (size_t)(2 * 16 * NODEF * sizeof(_Float16))
                 + 2 * (size_t)N_NODES * N_NODES * sizeof(float);
    gat_head_kernel<<<TOKENS / 64, 128, shmem, stream>>>(
        x, pW1, b1, pW2, b2, sA1, pWg1, bg1, sA2, pWg2, bg2, pWc, bc, out);
}
